// Transformer_47888885350564
// MI455X (gfx1250) — matrix operand order repaired and checked by reading, not compiled, not run on hardware
//
#include <hip/hip_runtime.h>
#include <hip/hip_bf16.h>

#define NN   32768
#define FEAT 512
#define HID  512
#define NH   4
#define HD   128
#define NB   621
#define NBP  640
#define NC   64

#define DN     0.29730177875068026f   // 128^-0.25 (folded into proj)
#define DIAGC  0.04419417382415922f   // 0.5 * 128^-0.5
#define RATIO  0.04012862117f         // 621^-0.5
#define FEPS   1e-4f

typedef __attribute__((ext_vector_type(16))) _Float16 v16h;
typedef __attribute__((ext_vector_type(8)))  float    v8f;
typedef unsigned int v4u __attribute__((ext_vector_type(4)));
typedef int          v8i __attribute__((ext_vector_type(8)));
typedef int          v4i __attribute__((ext_vector_type(4)));

#if __has_builtin(__builtin_amdgcn_global_load_async_to_lds_b128) && \
    __has_builtin(__builtin_amdgcn_s_wait_asynccnt)
#define HAVE_ASYNC_LDS 1
#else
#define HAVE_ASYNC_LDS 0
#endif

#if __has_builtin(__builtin_amdgcn_tensor_load_to_lds) && \
    __has_builtin(__builtin_amdgcn_s_wait_tensorcnt)
#define HAVE_TDM 1
#else
#define HAVE_TDM 0
#endif

union FragU { v16h v; unsigned int d[8]; };

// A operand: 16x32 f16 tile, row-major source (stride in elements).
// lanes 0-15 -> M=lane, K in {0..7,16..23}; lanes 16-31 -> M=lane-16, K in {8..15,24..31}
__device__ __forceinline__ v16h frag_a(const _Float16* base, int stride, int lane) {
  FragU f;
  int m = lane & 15, half = lane >> 4;
  const _Float16* rp = base + m * stride + half * 8;
#pragma unroll
  for (int i = 0; i < 4; ++i) f.d[i]     = *(const unsigned int*)(rp + 2 * i);
#pragma unroll
  for (int i = 0; i < 4; ++i) f.d[i + 4] = *(const unsigned int*)(rp + 16 + 2 * i);
  return f.v;
}

// B operand: 32x16 tile given as Bt[N][K] row-major.
// lanes 0-15 -> N=lane, K in {0..7,16..23}; lanes 16-31 -> N=lane-16, K in {8..15,24..31}
__device__ __forceinline__ v16h frag_b(const _Float16* base, int stride, int lane) {
  FragU f;
  int n = lane & 15, half = lane >> 4;
  const _Float16* rp = base + n * stride + half * 8;
#pragma unroll
  for (int i = 0; i < 4; ++i) f.d[i]     = *(const unsigned int*)(rp + 2 * i);
#pragma unroll
  for (int i = 0; i < 4; ++i) f.d[i + 4] = *(const unsigned int*)(rp + 16 + 2 * i);
  return f.v;
}

__device__ __forceinline__ v8f wmma16(v16h a, v16h b, v8f c) {
  return __builtin_amdgcn_wmma_f32_16x16x32_f16(false, a, false, b, (short)0, c, false, false);
}

__device__ __forceinline__ unsigned fflip(float f) {
  unsigned u = __float_as_uint(f);
  return (u & 0x80000000u) ? ~u : (u | 0x80000000u);
}
__device__ __forceinline__ float funflip(unsigned u) {
  unsigned v = (u & 0x80000000u) ? (u ^ 0x80000000u) : ~u;
  return __uint_as_float(v);
}

// ---------------- conversion kernels ----------------
__global__ void k_cvt(const float* __restrict__ src, _Float16* __restrict__ dst, int n) {
  int i = blockIdx.x * blockDim.x + threadIdx.x;
  if (i < n) dst[i] = (_Float16)src[i];
}

__global__ void k_cvt_proj(const float* __restrict__ proj, _Float16* __restrict__ dst) {
  int i = blockIdx.x * blockDim.x + threadIdx.x;     // over NBP*HD
  if (i >= NBP * HD) return;
  int r = i / HD, c = i - r * HD;
  float v = (r < NB) ? proj[r * HD + c] * DN : 0.f;  // fold data_normalizer, zero-pad
  dst[i] = (_Float16)v;
}

// ---------------- QKV GEMM: Out[n,512] = X @ W^T + b ----------------
// A panel (32 x 512 f16 = 32KB) staged once into LDS via async-to-LDS copies.
__global__ void k_gemm_qkv(const _Float16* __restrict__ A, const _Float16* __restrict__ Bt,
                           const float* __restrict__ bias, _Float16* __restrict__ Out) {
  __shared__ __align__(16) _Float16 Ash[32][FEAT];
  int block_m = blockIdx.x * 32;
  int wave = threadIdx.x >> 5, lane = threadIdx.x & 31;

  const _Float16* gsrc = A + (size_t)block_m * FEAT;   // contiguous 32KB panel
#if HAVE_ASYNC_LDS
  {
    const _Float16* g = gsrc + threadIdx.x * 8;        // 16B per chunk
    _Float16* l = &Ash[0][0] + threadIdx.x * 8;
#pragma unroll
    for (int i = 0; i < 8; ++i)
      __builtin_amdgcn_global_load_async_to_lds_b128(
          (__attribute__((address_space(1))) v4i*)(unsigned long long)(g + i * 2048),
          (__attribute__((address_space(3))) v4i*)(l + i * 2048), 0, 0);
    __builtin_amdgcn_s_wait_asynccnt(0);
  }
  __syncthreads();
#else
  for (int i = threadIdx.x; i < 32 * FEAT / 8; i += 256)
    ((float4*)&Ash[0][0])[i] = ((const float4*)gsrc)[i];
  __syncthreads();
#endif

  for (int t = wave; t < 64; t += 8) {
    int mt = t >> 5, nt = t & 31;
    v8f acc = {};
    const _Float16* a0 = &Ash[mt * 16][0];
    const _Float16* b0 = Bt + (size_t)(nt * 16) * FEAT;
    for (int k = 0; k < FEAT; k += 32)
      acc = wmma16(frag_a(a0 + k, FEAT, lane), frag_b(b0 + k, FEAT, lane), acc);
    int col = nt * 16 + (lane & 15);
    int row = block_m + mt * 16 + (lane >> 4) * 8;
    float bb = bias[col];
#pragma unroll
    for (int r = 0; r < 8; ++r)
      Out[(size_t)(row + r) * HID + col] = (_Float16)(acc[r] + bb);
  }
}

// ---------------- diag = 0.5 * d^-0.5 * sum(q^2) per (head,node) ----------------
__global__ void k_diag(const _Float16* __restrict__ Q, float* __restrict__ diag) {
  int i = blockIdx.x * blockDim.x + threadIdx.x;     // over NN*NH
  if (i >= NN * NH) return;
  int n = i >> 2, h = i & 3;
  const _Float16* p = Q + (size_t)n * HID + h * HD;
  float s = 0.f;
#pragma unroll 8
  for (int k = 0; k < HD; ++k) { float v = (float)p[k]; s += v * v; }
  diag[h * NN + n] = s * DIAGC;
}

// ---------------- k-side global max of data_dash (stab for keys) ----------------
__global__ void k_kmax(const _Float16* __restrict__ K, const _Float16* __restrict__ P,
                       unsigned* __restrict__ kmaxb) {
  const int nchunks = NN / 32;
  int h = blockIdx.x / nchunks, chunk = blockIdx.x - h * nchunks;
  int row0 = chunk * 32;
  int wave = threadIdx.x >> 5, lane = threadIdx.x & 31;
  float lmax = -3.4e38f;
  for (int nt = wave; nt < NBP / 16; nt += 8) {
    int col = nt * 16 + (lane & 15);
    for (int mt = 0; mt < 2; ++mt) {
      v8f acc = {};
      const _Float16* a0 = K + (size_t)(row0 + mt * 16) * HID + h * HD;
      const _Float16* b0 = P + (size_t)(nt * 16) * HD;
      for (int k = 0; k < HD; k += 32)
        acc = wmma16(frag_a(a0 + k, HID, lane), frag_b(b0 + k, HD, lane), acc);
      if (col < NB) {
#pragma unroll
        for (int r = 0; r < 8; ++r) lmax = fmaxf(lmax, acc[r]);
      }
    }
  }
  __shared__ float red[256];
  red[threadIdx.x] = lmax; __syncthreads();
  for (int s = 128; s > 0; s >>= 1) {
    if ((int)threadIdx.x < s) red[threadIdx.x] = fmaxf(red[threadIdx.x], red[threadIdx.x + s]);
    __syncthreads();
  }
  if (threadIdx.x == 0) atomicMax(&kmaxb[h], fflip(red[0]));
}

// ---------------- fused k' -> k_cumsum + context ----------------
// K tile (32 rows x 128 f16) staged into LDS by the Tensor Data Mover (wave 0),
// then re-used by 40 N-tiles x 2 M-tiles of WMMA.
__global__ void k_kctx(const _Float16* __restrict__ K, const _Float16* __restrict__ V,
                       const _Float16* __restrict__ P, const float* __restrict__ diagK,
                       const unsigned* __restrict__ kmaxb,
                       float* __restrict__ kcs, float* __restrict__ ctx) {
  __shared__ __align__(16) _Float16 Ksh[32][HD];   // staged K tile  (8KB)
  __shared__ __align__(16) _Float16 kT[NBP][32];   // k' transposed: [m][node] (40KB)
  __shared__ __align__(16) _Float16 vT[HD][32];    // v transposed:  [e][node] (8KB)
  const int nchunks = NN / 32;
  int h = blockIdx.x / nchunks, chunk = blockIdx.x - h * nchunks;
  int row0 = chunk * 32;
  int wave = threadIdx.x >> 5, lane = threadIdx.x & 31;
  float stab = funflip(kmaxb[h]);

  const _Float16* gk = K + (size_t)row0 * HID + h * HD;
#if HAVE_TDM
  if (wave == 0) {
    // D# for a 2D tile: data_size=8B units; tile 32x32 units (256B rows x 32 rows);
    // tensor row pitch = 128 units (1024B = HID f16); LDS dest = Ksh (packed).
    unsigned long long ga = (unsigned long long)(const void*)gk;
    unsigned lds_off = (unsigned)(unsigned long long)
        (__attribute__((address_space(3))) _Float16*)&Ksh[0][0];
    v4u g0 = { 1u,                                   // count=1, user descriptor
               lds_off,                               // lds_addr (bytes)
               (unsigned)(ga & 0xffffffffu),          // global_addr[31:0]
               (unsigned)((ga >> 32) & 0x01ffffffu) | 0x80000000u }; // [56:32] | type=2
    v8i g1 = { (int)0x00030000,                       // data_size=3 (8B)
               (int)(128u << 16),                     // tensor_dim0 = 128 units
               (int)(32u  << 16),                     // tensor_dim1 = 32 rows
               (int)(32u  << 16),                     // tile_dim0   = 32 units (256B)
               (int)32,                               // tile_dim1   = 32 rows
               (int)128,                              // tensor_dim0_stride = 128 units
               0, 0 };
    v4i gz  = { 0, 0, 0, 0 };
    v8i gz8 = { 0, 0, 0, 0, 0, 0, 0, 0 };
    __builtin_amdgcn_tensor_load_to_lds(g0, g1, gz, gz, gz8, 0);
    __builtin_amdgcn_s_wait_tensorcnt(0);
  }
#else
  for (int i = threadIdx.x; i < 32 * HD / 8; i += 256) {
    int r = i >> 4, c = (i & 15) * 8;                 // 8 f16 per chunk
    *(float4*)&Ksh[r][c] = *(const float4*)(gk + (size_t)r * HID + c);
  }
#endif

  for (int i = threadIdx.x; i < HD * 32; i += 256) {
    int e = i >> 5, j = i & 31;
    vT[e][j] = V[(size_t)(row0 + j) * HID + h * HD + e];
  }
  __syncthreads();

  for (int nt = wave; nt < NBP / 16; nt += 8) {
    int col = nt * 16 + (lane & 15);
    for (int mt = 0; mt < 2; ++mt) {
      v8f acc = {};
      const _Float16* a0 = &Ksh[mt * 16][0];
      const _Float16* b0 = P + (size_t)(nt * 16) * HD;
      for (int k = 0; k < HD; k += 32)
        acc = wmma16(frag_a(a0 + k, HD, lane), frag_b(b0 + k, HD, lane), acc);
      int rl0 = mt * 16 + (lane >> 4) * 8;
      float csum = 0.f;
#pragma unroll
      for (int r = 0; r < 8; ++r) {
        int rl = rl0 + r;
        float kp = 0.f;
        if (col < NB)
          kp = RATIO * (__expf(acc[r] - diagK[h * NN + row0 + rl] - stab) + FEPS);
        kT[col][rl] = (_Float16)kp;
        csum += kp;
      }
      if (col < NB) atomicAdd(&kcs[h * NBP + col], csum);
    }
  }
  __syncthreads();

  // context[m,e] += k'^T @ v  (M=640, N=128, K=32; wave owns one N-tile)
  for (int mt = 0; mt < NBP / 16; ++mt) {
    v8f acc = {};
    acc = wmma16(frag_a(&kT[mt * 16][0], 32, lane), frag_b(&vT[wave * 16][0], 32, lane), acc);
    int e  = wave * 16 + (lane & 15);
    int m0 = mt * 16 + (lane >> 4) * 8;
#pragma unroll
    for (int r = 0; r < 8; ++r)
      atomicAdd(&ctx[(size_t)(h * NBP + m0 + r) * HD + e], acc[r]);
  }
}

// ---------------- ctx f32 [h][m][e] -> ctxT f16 [h][e][m] ----------------
__global__ void k_ctxT(const float* __restrict__ ctx, _Float16* __restrict__ ctxT) {
  int i = blockIdx.x * blockDim.x + threadIdx.x;
  if (i >= NH * NBP * HD) return;
  int h = i / (NBP * HD);
  int rem = i - h * (NBP * HD);
  int e = rem / NBP, m = rem - e * NBP;
  ctxT[i] = (_Float16)ctx[(size_t)(h * NBP + m) * HD + e];
}

// ---------------- fused q' -> d_inv -> out tile -> relu ----------------
#define QCH 16
__global__ void k_qout(const _Float16* __restrict__ Q, const _Float16* __restrict__ P,
                       const float* __restrict__ diagQ, const float* __restrict__ kcs,
                       const _Float16* __restrict__ ctxT, _Float16* __restrict__ hrel) {
  __shared__ __align__(16) float    dd[QCH][NBP + 8];
  __shared__ __align__(16) _Float16 qp[QCH][NBP];
  __shared__ float rmax[QCH], dinv[QCH];
  const int nchunks = NN / QCH;
  int h = blockIdx.x / nchunks, chunk = blockIdx.x - h * nchunks;
  int row0 = chunk * QCH;
  int wave = threadIdx.x >> 5, lane = threadIdx.x & 31;

  for (int nt = wave * 5; nt < wave * 5 + 5; ++nt) {   // 8 waves * 5 = 40 N-tiles
    v8f acc = {};
    const _Float16* a0 = Q + (size_t)row0 * HID + h * HD;
    const _Float16* b0 = P + (size_t)(nt * 16) * HD;
    for (int k = 0; k < HD; k += 32) {
      __builtin_prefetch(b0 + k + 256, 0, 1);
      acc = wmma16(frag_a(a0 + k, HID, lane), frag_b(b0 + k, HD, lane), acc);
    }
    int col = nt * 16 + (lane & 15);
    int rl0 = (lane >> 4) * 8;
#pragma unroll
    for (int r = 0; r < 8; ++r) dd[rl0 + r][col] = acc[r];
  }
  __syncthreads();

  if (threadIdx.x < QCH) {
    float m = -3.4e38f;
    for (int c = 0; c < NB; ++c) m = fmaxf(m, dd[threadIdx.x][c]);
    rmax[threadIdx.x] = m;
  }
  __syncthreads();

  for (int i = threadIdx.x; i < QCH * NBP; i += 256) {
    int r = i / NBP, c = i - r * NBP;
    float q = 0.f;
    if (c < NB)
      q = RATIO * (__expf(dd[r][c] - diagQ[h * NN + row0 + r] - rmax[r]) + FEPS);
    qp[r][c] = (_Float16)q;
  }
  __syncthreads();

  if (threadIdx.x < QCH) {
    float s = 0.f;
    for (int c = 0; c < NB; ++c) s += (float)qp[threadIdx.x][c] * kcs[h * NBP + c];
    dinv[threadIdx.x] = 1.f / s;
  }
  __syncthreads();

  // out[n,e] = q' @ ctx ; M=16, N=128 (wave owns one N-tile), K=640
  {
    v8f acc = {};
    const _Float16* b0 = ctxT + (size_t)(h * HD + wave * 16) * NBP;
    for (int k = 0; k < NBP; k += 32)
      acc = wmma16(frag_a(&qp[0][k], NBP, lane), frag_b(b0 + k, NBP, lane), acc);
    int e   = wave * 16 + (lane & 15);
    int rl0 = (lane >> 4) * 8;
#pragma unroll
    for (int r = 0; r < 8; ++r) {
      float v = acc[r] * dinv[rl0 + r];
      v = v > 0.f ? v : 0.f;                              // relu
      hrel[(size_t)(row0 + rl0 + r) * HID + h * HD + e] = (_Float16)v;
    }
  }
}

// ---------------- classifier: logits = hrel @ W1^T + b1 ----------------
__global__ void k_logits(const _Float16* __restrict__ Hr, const _Float16* __restrict__ W1h,
                         const float* __restrict__ b1, float* __restrict__ logits) {
  int block_m = blockIdx.x * 64;
  int wave = threadIdx.x >> 5, lane = threadIdx.x & 31;
  for (int t = wave; t < 16; t += 8) {
    int mt = t >> 2, nt = t & 3;
    v8f acc = {};
    const _Float16* a0 = Hr + (size_t)(block_m + mt * 16) * HID;
    const _Float16* b0 = W1h + (size_t)(nt * 16) * HID;
    for (int k = 0; k < HID; k += 32) {
      __builtin_prefetch(a0 + k + 128, 0, 1);
      acc = wmma16(frag_a(a0 + k, HID, lane), frag_b(b0 + k, HID, lane), acc);
    }
    int col = nt * 16 + (lane & 15);
    int row = block_m + mt * 16 + (lane >> 4) * 8;
    float bb = b1[col];
#pragma unroll
    for (int r = 0; r < 8; ++r)
      logits[(size_t)(row + r) * NC + col] = acc[r] + bb;
  }
}

__global__ void k_lsm_row(const float* __restrict__ logits, float* __restrict__ out0) {
  int n = blockIdx.x * blockDim.x + threadIdx.x;
  if (n >= NN) return;
  const float* p = logits + (size_t)n * NC;
  float m = -3.4e38f;
  for (int c = 0; c < NC; ++c) m = fmaxf(m, p[c]);
  float s = 0.f;
  for (int c = 0; c < NC; ++c) s += __expf(p[c] - m);
  float ls = __logf(s);
  for (int c = 0; c < NC; ++c) out0[(size_t)n * NC + c] = p[c] - m - ls;
}

__global__ void k_sm_col(const float* __restrict__ logits, float* __restrict__ out2) {
  int c = blockIdx.x;
  __shared__ float red[256];
  float m = -3.4e38f;
  for (int n = threadIdx.x; n < NN; n += 256) m = fmaxf(m, logits[(size_t)n * NC + c]);
  red[threadIdx.x] = m; __syncthreads();
  for (int s = 128; s > 0; s >>= 1) {
    if ((int)threadIdx.x < s) red[threadIdx.x] = fmaxf(red[threadIdx.x], red[threadIdx.x + s]);
    __syncthreads();
  }
  m = red[0]; __syncthreads();
  float sum = 0.f;
  for (int n = threadIdx.x; n < NN; n += 256) sum += __expf(logits[(size_t)n * NC + c] - m);
  red[threadIdx.x] = sum; __syncthreads();
  for (int s = 128; s > 0; s >>= 1) {
    if ((int)threadIdx.x < s) red[threadIdx.x] += red[threadIdx.x + s];
    __syncthreads();
  }
  float inv = 1.f / red[0];
  for (int n = threadIdx.x; n < NN; n += 256)
    out2[(size_t)n * NC + c] = __expf(logits[(size_t)n * NC + c] - m) * inv;
}

extern "C" void kernel_launch(void* const* d_in, const int* in_sizes, int n_in,
                              void* d_out, int out_size, void* d_ws, size_t ws_size,
                              hipStream_t stream) {
  const float* x    = (const float*)d_in[0];
  /* edge_index d_in[1] unused by the module */
  const float* Wq   = (const float*)d_in[2];
  const float* bq   = (const float*)d_in[3];
  const float* Wk   = (const float*)d_in[4];
  const float* bk   = (const float*)d_in[5];
  const float* Wv   = (const float*)d_in[6];
  const float* bv   = (const float*)d_in[7];
  const float* W1   = (const float*)d_in[8];
  const float* b1   = (const float*)d_in[9];
  const float* proj = (const float*)d_in[10];
  (void)in_sizes; (void)n_in; (void)out_size; (void)ws_size;

  char* base = (char*)d_ws;
  size_t off = 0;
  auto alloc = [&](size_t bytes) { size_t r = off; off = (off + bytes + 255) & ~(size_t)255; return r; };

  _Float16* Xh   = (_Float16*)(base + alloc((size_t)NN * FEAT * 2));
  _Float16* Wqh  = (_Float16*)(base + alloc((size_t)HID * FEAT * 2));
  _Float16* Wkh  = (_Float16*)(base + alloc((size_t)HID * FEAT * 2));
  _Float16* Wvh  = (_Float16*)(base + alloc((size_t)HID * FEAT * 2));
  _Float16* W1h  = (_Float16*)(base + alloc((size_t)NC * HID * 2));
  _Float16* Pjh  = (_Float16*)(base + alloc((size_t)NBP * HD * 2));
  _Float16* Qh   = (_Float16*)(base + alloc((size_t)NN * HID * 2));
  _Float16* Kh   = (_Float16*)(base + alloc((size_t)NN * HID * 2));
  _Float16* Vh   = (_Float16*)(base + alloc((size_t)NN * HID * 2));
  float*    dQ   = (float*)   (base + alloc((size_t)NH * NN * 4));
  float*    dK   = (float*)   (base + alloc((size_t)NH * NN * 4));
  unsigned* kmx  = (unsigned*)(base + alloc(256));
  float*    kcs  = (float*)   (base + alloc((size_t)NH * NBP * 4));
  float*    ctx  = (float*)   (base + alloc((size_t)NH * NBP * HD * 4));
  _Float16* ctxT = (_Float16*)(base + alloc((size_t)NH * HD * NBP * 2));
  _Float16* hrel = (_Float16*)(base + alloc((size_t)NN * HID * 2));

  float* out0 = (float*)d_out;                       // log_softmax(logits, axis=1)
  float* lgts = out0 + (size_t)NN * NC;              // logits
  float* out2 = lgts + (size_t)NN * NC;              // softmax(logits, axis=0)

  // zero accumulation state (must happen every call: graphed replays)
  (void)hipMemsetAsync(kmx, 0, 256, stream);
  (void)hipMemsetAsync(kcs, 0, (size_t)NH * NBP * 4, stream);
  (void)hipMemsetAsync(ctx, 0, (size_t)NH * NBP * HD * 4, stream);

  // conversions to f16
  k_cvt<<<(NN * FEAT + 255) / 256, 256, 0, stream>>>(x,  Xh,  NN * FEAT);
  k_cvt<<<(HID * FEAT + 255) / 256, 256, 0, stream>>>(Wq, Wqh, HID * FEAT);
  k_cvt<<<(HID * FEAT + 255) / 256, 256, 0, stream>>>(Wk, Wkh, HID * FEAT);
  k_cvt<<<(HID * FEAT + 255) / 256, 256, 0, stream>>>(Wv, Wvh, HID * FEAT);
  k_cvt<<<(NC * HID + 255) / 256, 256, 0, stream>>>(W1, W1h, NC * HID);
  k_cvt_proj<<<(NBP * HD + 255) / 256, 256, 0, stream>>>(proj, Pjh);

  // QKV projections
  k_gemm_qkv<<<NN / 32, 256, 0, stream>>>(Xh, Wqh, bq, Qh);
  k_gemm_qkv<<<NN / 32, 256, 0, stream>>>(Xh, Wkh, bk, Kh);
  k_gemm_qkv<<<NN / 32, 256, 0, stream>>>(Xh, Wvh, bv, Vh);

  // diag terms
  k_diag<<<(NN * NH + 255) / 256, 256, 0, stream>>>(Qh, dQ);
  k_diag<<<(NN * NH + 255) / 256, 256, 0, stream>>>(Kh, dK);

  // key-side stabilizer (global max of data_dash per head)
  k_kmax<<<NH * (NN / 32), 256, 0, stream>>>(Kh, Pjh, kmx);

  // fused k' -> k_cumsum + context (TDM-staged K tiles)
  k_kctx<<<NH * (NN / 32), 256, 0, stream>>>(Kh, Vh, Pjh, dK, kmx, kcs, ctx);
  k_ctxT<<<(NH * NBP * HD + 255) / 256, 256, 0, stream>>>(ctx, ctxT);

  // fused q' -> attention output -> relu
  k_qout<<<NH * (NN / QCH), 256, 0, stream>>>(Qh, Pjh, dQ, kcs, ctxT, hrel);

  // classifier + softmaxes
  k_logits<<<NN / 64, 256, 0, stream>>>(hrel, W1h, b1, lgts);
  k_lsm_row<<<NN / 256, 256, 0, stream>>>(lgts, out0);
  k_sm_col<<<NC, 256, 0, stream>>>(lgts, out2);
}
